// PointNet2_59296318488826
// MI455X (gfx1250) — matrix operand order repaired and checked by reading, compiled, not run on hardware
//
#include <hip/hip_runtime.h>
#include <hip/hip_bf16.h>

typedef __attribute__((ext_vector_type(16))) __bf16 v16bf;
typedef __attribute__((ext_vector_type(8)))  float  v8f;

// Packed operand layouts (bf16):
//  A (weights): Wp[((mt*K32 + kt)*32 + lane)*16 + e], e maps K = kt*32 + hi*16 + e, m = mt*16 + lane%16
//  B/X:         Xp[((kt*N + n)*32 + j], j = k&31  (lane reads 16 contiguous at j = hi*16)
static __device__ __forceinline__ size_t xpos(int row, int col, int N) {
    return ((size_t)(row >> 5) * N + col) * 32 + (row & 31);
}

// ---------------------------------------------------------------------------
// GEMM: Y = W * X. Wave tile = 32(M) x 64(N): 2 A-frags x 4 B-frags -> 8 WMMAs
// per 32-deep K step. Requires N % 64 == 0 (true for every launch here).
// epi==0: relu((y+cb)*sc+bb); epi==1: y+cb. Yf (f32 [M,N]) / Yp (packed) optional.
// ---------------------------------------------------------------------------
__global__ void gemm_bn_kernel(const __bf16* __restrict__ Ap, const __bf16* __restrict__ Xp,
                               float* __restrict__ Yf, __bf16* __restrict__ Yp,
                               const float* __restrict__ cb, const float* __restrict__ sc,
                               const float* __restrict__ bb,
                               int M, int K32, int N, int epi)
{
    const int lane = threadIdx.x & 31;
    const int wid  = threadIdx.x >> 5;
    const int nt   = blockIdx.x * (blockDim.x >> 5) + wid;  // 64-col tile index
    const int n0   = nt * 64;
    if (n0 >= N) return;                     // uniform per wave
    const int mt0  = blockIdx.y * 2;         // two 16-row A tiles (pack is even-padded)
    const int lm = lane & 15;
    const int hi = lane >> 4;

    const __bf16* aP0 = Ap + ((size_t)mt0       * K32 * 32 + lane) * 16;
    const __bf16* aP1 = Ap + ((size_t)(mt0 + 1) * K32 * 32 + lane) * 16;
    const __bf16* xP  = Xp + ((size_t)(n0 + lm)) * 32 + hi * 16;

    v8f zero = {0.f,0.f,0.f,0.f,0.f,0.f,0.f,0.f};
    v8f acc[2][4];
#pragma unroll
    for (int mi = 0; mi < 2; ++mi)
#pragma unroll
        for (int ni = 0; ni < 4; ++ni) acc[mi][ni] = zero;

    for (int kt = 0; kt < K32; ++kt) {
        v16bf a0 = *(const v16bf*)(aP0 + (size_t)kt * 512);
        v16bf a1 = *(const v16bf*)(aP1 + (size_t)kt * 512);
        const __bf16* xb = xP + (size_t)kt * N * 32;
#pragma unroll
        for (int ni = 0; ni < 4; ++ni) {
            v16bf b = *(const v16bf*)(xb + ni * 16 * 32);
            acc[0][ni] = __builtin_amdgcn_wmma_f32_16x16x32_bf16(false, a0, false, b, (short)0,
                                                                 acc[0][ni], false, false);
            acc[1][ni] = __builtin_amdgcn_wmma_f32_16x16x32_bf16(false, a1, false, b, (short)0,
                                                                 acc[1][ni], false, false);
        }
    }
#pragma unroll
    for (int mi = 0; mi < 2; ++mi) {
        int m0 = (mt0 + mi) * 16;
#pragma unroll
        for (int r = 0; r < 8; ++r) {
            int mm = m0 + r + hi * 8;
            if (mm >= M) continue;
            float add   = cb[mm];
            float scale = (epi == 0) ? sc[mm] : 1.0f;
            float shift = (epi == 0) ? bb[mm] : 0.0f;
#pragma unroll
            for (int ni = 0; ni < 4; ++ni) {
                float v = acc[mi][ni][r];
                v = (v + add) * scale + shift;
                if (epi == 0) v = fmaxf(v, 0.0f);
                int nn = n0 + ni * 16 + lm;
                if (Yf) Yf[(size_t)mm * N + nn] = v;
                if (Yp) Yp[xpos(mm, nn, N)] = (__bf16)v;
            }
        }
    }
}

// ---------------------------------------------------------------------------
// Pack weights [M,K] f32 -> WMMA A-fragment bf16 layout, MtE 16-row tiles
// (MtE even, zero-padded in both M and K).
// ---------------------------------------------------------------------------
__global__ void pack_w_kernel(const float* __restrict__ W, __bf16* __restrict__ Wp,
                              int M, int K, int K32, int MtE)
{
    int i = blockIdx.x * blockDim.x + threadIdx.x;
    int total = MtE * K32 * 512;
    if (i >= total) return;
    int e    = i & 15;
    int lane = (i >> 4) & 31;
    int kt   = (i >> 9) % K32;
    int mt   = i / (K32 * 512);
    int lm = lane & 15, hi = lane >> 4;
    int m  = mt * 16 + lm;
    int ka = kt * 32 + hi * 16 + e;
    float v = (m < M && ka < K) ? W[(size_t)m * K + ka] : 0.0f;
    Wp[i] = (__bf16)v;
}

__global__ void zero_rows_kernel(__bf16* __restrict__ Xp, int N, int r0, int r1)
{
    int i = blockIdx.x * blockDim.x + threadIdx.x;
    int rows = r1 - r0;
    if (i >= rows * N) return;
    int row = r0 + i / N, col = i - (i / N) * N;
    Xp[xpos(row, col, N)] = (__bf16)0.0f;
}

// ---------------------------------------------------------------------------
// Farthest point sampling: one block per batch. Writes centers point-major;
// optionally packed-bf16 xyz rows 0..2 of a GEMM input matrix.
// ---------------------------------------------------------------------------
template<int PER>
__global__ void fps_kernel(const float* __restrict__ src, int ss, int n, int npoint,
                           float* __restrict__ pm, __bf16* __restrict__ cm, int cmLd)
{
    const int b = blockIdx.x;
    const int tid = threadIdx.x;
    const float* p = src + (size_t)b * n * ss;
    float* opm = pm + (size_t)b * npoint * 3;
    float dist[PER];
#pragma unroll
    for (int j = 0; j < PER; ++j) dist[j] = 1e10f;
    __shared__ float s_v[256]; __shared__ int s_i[256];
    __shared__ float s_c[3];   __shared__ int s_far;
    int far = 0;
    for (int it = 0; it < npoint; ++it) {
        if (tid == 0) {
            float x = p[(size_t)far*ss+0], y = p[(size_t)far*ss+1], z = p[(size_t)far*ss+2];
            s_c[0]=x; s_c[1]=y; s_c[2]=z;
            opm[it*3+0]=x; opm[it*3+1]=y; opm[it*3+2]=z;
            if (cm) {
                int col = b*npoint + it;
                cm[xpos(0, col, cmLd)] = (__bf16)x;
                cm[xpos(1, col, cmLd)] = (__bf16)y;
                cm[xpos(2, col, cmLd)] = (__bf16)z;
            }
        }
        __syncthreads();
        float fx=s_c[0], fy=s_c[1], fz=s_c[2];
        float bv = -1.0f; int bi = 0;
#pragma unroll
        for (int j = 0; j < PER; ++j) {
            int pt = tid + 256 * j;
            float dx = p[(size_t)pt*ss+0]-fx, dy = p[(size_t)pt*ss+1]-fy, dz = p[(size_t)pt*ss+2]-fz;
            float d = dx*dx+dy*dy+dz*dz;
            float nd = fminf(dist[j], d);
            dist[j] = nd;
            if (nd > bv) { bv = nd; bi = pt; }
        }
        s_v[tid]=bv; s_i[tid]=bi;
        __syncthreads();
        for (int off=128; off>0; off>>=1) {
            if (tid < off) {
                float ov = s_v[tid+off]; int oi = s_i[tid+off];
                if (ov > s_v[tid] || (ov == s_v[tid] && oi < s_i[tid])) { s_v[tid]=ov; s_i[tid]=oi; }
            }
            __syncthreads();
        }
        if (tid==0) s_far = s_i[0];
        __syncthreads();
        far = s_far;
    }
}

// ---------------------------------------------------------------------------
// Ball query + grouping -> packed bf16 GEMM input (rows 0..Cf-1 feats,
// Cf..Cf+2 rel-xyz, Cf+3..rowsP-1 zero pad).
// ---------------------------------------------------------------------------
__global__ void group_kernel(const float* __restrict__ xyz, int sx,
                             const float* __restrict__ feat, int sf, int Cf,
                             const float* __restrict__ ctr,
                             int n, int K, float r2,
                             __bf16* __restrict__ Xp, int ld, int rowsP)
{
    const int s = blockIdx.x;
    const int tid = threadIdx.x;
    const int wid = tid >> 5, lane = tid & 31;
    __shared__ int s_wcnt[8];
    __shared__ int s_nbr[128];
    const float cx = ctr[s*3+0], cy = ctr[s*3+1], cz = ctr[s*3+2];
    int base = 0;
    for (int c0 = 0; c0 < n; c0 += 256) {
        int j = c0 + tid;
        bool pred = false;
        if (j < n) {
            float dx = xyz[(size_t)j*sx+0]-cx, dy = xyz[(size_t)j*sx+1]-cy, dz = xyz[(size_t)j*sx+2]-cz;
            pred = (dx*dx+dy*dy+dz*dz) <= r2;
        }
        unsigned mask = (unsigned)__ballot(pred);
        if (lane == 0) s_wcnt[wid] = __popc(mask);
        __syncthreads();
        int pre = 0, tot = 0;
#pragma unroll
        for (int w = 0; w < 8; ++w) { int c = s_wcnt[w]; tot += c; if (w < wid) pre += c; }
        if (pred) {
            int slot = base + pre + __popc(mask & ((1u << lane) - 1u));
            if (slot < K) s_nbr[slot] = j;
        }
        base += tot;
        __syncthreads();
        if (base >= K) break;
    }
    int cnt = base < K ? base : K;
    if (tid == 0 && cnt == 0) s_nbr[0] = n - 1;
    __syncthreads();
    int padj = s_nbr[0];
    for (int k = tid; k < K; k += 256) if (k >= cnt && k > 0) s_nbr[k] = padj;
    __syncthreads();
    const int rows = Cf + 3;
    const int tote = K * rowsP;
    for (int e = tid; e < tote; e += 256) {
        int k = e / rowsP, c = e - k * rowsP;
        int j = s_nbr[k];
        int col = s * K + k;
        float v = 0.0f;
        if (c < Cf) v = feat[(size_t)j * sf + c];
        else if (c < rows) { int d = c - Cf; v = xyz[(size_t)j * sx + d] - ctr[s*3 + d]; }
        Xp[xpos(c, col, ld)] = (__bf16)v;
    }
}

// ---------------------------------------------------------------------------
// Max-pool over K; writes f32 point-major + up to two packed-bf16 dests.
// ---------------------------------------------------------------------------
__global__ void pool_kernel(const float* __restrict__ Y, int D, int S, int K,
                            float* __restrict__ pm, int pmStride, int pmColBase, int pmOff,
                            __bf16* __restrict__ cm1, int ldc1, int c1off, int cb1,
                            __bf16* __restrict__ cm2, int ldc2, int c2off, int cb2)
{
    int i = blockIdx.x * blockDim.x + threadIdx.x;
    if (i >= S * D) return;
    int s = i / D, d = i - s * D;
    const float* y = Y + (size_t)d * S * K + (size_t)s * K;
    float m = -3.0e38f;
    for (int k = 0; k < K; ++k) m = fmaxf(m, y[k]);
    pm[(size_t)(pmColBase + s) * pmStride + pmOff + d] = m;
    if (cm1) cm1[xpos(c1off + d, cb1 + s, ldc1)] = (__bf16)m;
    if (cm2) cm2[xpos(c2off + d, cb2 + s, ldc2)] = (__bf16)m;
}

__global__ void pool_all_kernel(const float* __restrict__ Y, float* __restrict__ l3)
{   // Y [1024, 512] (4 batches x 128 cols) -> l3[b*1024+d]
    int i = blockIdx.x * blockDim.x + threadIdx.x;
    if (i >= 4*1024) return;
    int b = i >> 10, d = i & 1023;
    const float* y = Y + (size_t)d*512 + b*128;
    float m = -3.0e38f;
    for (int j = 0; j < 128; ++j) m = fmaxf(m, y[j]);
    l3[i] = m;
}

__global__ void bcast_fp3_kernel(const float* __restrict__ l3, __bf16* __restrict__ X)
{   // X_fp3 rows 640..1663 = broadcast of l3 over each batch's 128 cols
    int i = blockIdx.x * blockDim.x + threadIdx.x;
    if (i >= 1024*512) return;
    int c = i >> 9, col = i & 511, b = col >> 7;
    X[xpos(640 + c, col, 512)] = (__bf16)l3[b*1024 + c];
}

__global__ void transpose_kernel(const float* __restrict__ cm, float* __restrict__ pm,
                                 int C, int Ncols)
{
    int i = blockIdx.x * blockDim.x + threadIdx.x;
    if (i >= C * Ncols) return;
    int col = i / C, c = i - col * C;
    pm[(size_t)col * C + c] = cm[(size_t)c * Ncols + col];
}

// ---------------------------------------------------------------------------
// FP2 interpolation: 3-NN among 128 l2 centers -> packed rows 320..575 of X_fp2.
// ---------------------------------------------------------------------------
__global__ void interp_fp2_kernel(const float* __restrict__ x1, const float* __restrict__ x2,
                                  const float* __restrict__ f2, __bf16* __restrict__ X)
{
    const int col = blockIdx.x;          // 0..2047
    const int b = col >> 9;
    const int tid = threadIdx.x;         // 128 threads
    __shared__ float s_d[128];
    __shared__ float s_v[128]; __shared__ int s_ix[128];
    __shared__ int s_sel[3];
    float cx = x1[col*3+0], cy = x1[col*3+1], cz = x1[col*3+2];
    {
        int j = b*128 + tid;
        float dx = x2[j*3+0]-cx, dy = x2[j*3+1]-cy, dz = x2[j*3+2]-cz;
        s_d[tid] = dx*dx+dy*dy+dz*dz;
    }
    __syncthreads();
    for (int r = 0; r < 3; ++r) {
        float v = s_d[tid];
        for (int qq = 0; qq < r; ++qq) if (tid == s_sel[qq]) v = 3.0e38f;
        s_v[tid] = v; s_ix[tid] = tid;
        __syncthreads();
        for (int off = 64; off > 0; off >>= 1) {
            if (tid < off) {
                float ov = s_v[tid+off]; int oi = s_ix[tid+off];
                if (ov < s_v[tid] || (ov == s_v[tid] && oi < s_ix[tid])) { s_v[tid]=ov; s_ix[tid]=oi; }
            }
            __syncthreads();
        }
        if (tid == 0) s_sel[r] = s_ix[0];
        __syncthreads();
    }
    int j0 = s_sel[0], j1 = s_sel[1], j2 = s_sel[2];
    float w0 = 1.0f/(s_d[j0]+1e-8f), w1 = 1.0f/(s_d[j1]+1e-8f), w2 = 1.0f/(s_d[j2]+1e-8f);
    float ws = w0+w1+w2; w0/=ws; w1/=ws; w2/=ws;
    const float* f0 = f2 + (size_t)(b*128 + j0) * 256;
    const float* f1 = f2 + (size_t)(b*128 + j1) * 256;
    const float* fB = f2 + (size_t)(b*128 + j2) * 256;
    for (int c = tid; c < 256; c += 128)
        X[xpos(320 + c, col, 2048)] = (__bf16)(w0*f0[c] + w1*f1[c] + w2*fB[c]);
}

// ---------------------------------------------------------------------------
// FP1 build: packed rows 0..5 = (coords,normals); 6..133 = 3-NN interp; 134..159 = 0.
// ---------------------------------------------------------------------------
__global__ void build_fp1_kernel(const float* __restrict__ in6, const float* __restrict__ x2,
                                 const float* __restrict__ f1, __bf16* __restrict__ X)
{
    const int col = blockIdx.x;   // 0..32767
    const int b = col >> 13;
    const int i = col & 8191;
    const int tid = threadIdx.x;  // 256
    __shared__ float s_d[512];
    __shared__ float s_v[256]; __shared__ int s_ix[256];
    __shared__ int s_sel[3];
    const float* pc = in6 + (size_t)(b*8192 + i) * 6;
    float cx = pc[0], cy = pc[1], cz = pc[2];
    for (int j = tid; j < 512; j += 256) {
        const float* qp = x2 + (size_t)(b*512 + j) * 3;
        float dx=qp[0]-cx, dy=qp[1]-cy, dz=qp[2]-cz;
        s_d[j] = dx*dx+dy*dy+dz*dz;
    }
    __syncthreads();
    for (int r = 0; r < 3; ++r) {
        int i0 = tid, i1 = tid + 256;
        float v0 = s_d[i0], v1 = s_d[i1];
        for (int qq = 0; qq < r; ++qq) { int sq = s_sel[qq]; if (i0 == sq) v0 = 3.0e38f; if (i1 == sq) v1 = 3.0e38f; }
        float v; int ix;
        if (v1 < v0) { v = v1; ix = i1; } else { v = v0; ix = i0; }
        s_v[tid] = v; s_ix[tid] = ix;
        __syncthreads();
        for (int off = 128; off > 0; off >>= 1) {
            if (tid < off) {
                float ov = s_v[tid+off]; int oi = s_ix[tid+off];
                if (ov < s_v[tid] || (ov == s_v[tid] && oi < s_ix[tid])) { s_v[tid]=ov; s_ix[tid]=oi; }
            }
            __syncthreads();
        }
        if (tid == 0) s_sel[r] = s_ix[0];
        __syncthreads();
    }
    int j0=s_sel[0], j1=s_sel[1], j2=s_sel[2];
    float w0=1.0f/(s_d[j0]+1e-8f), w1=1.0f/(s_d[j1]+1e-8f), w2=1.0f/(s_d[j2]+1e-8f);
    float ws=w0+w1+w2; w0/=ws; w1/=ws; w2/=ws;
    if (tid < 6)  X[xpos(tid, col, 32768)] = (__bf16)pc[tid];
    if (tid < 26) X[xpos(134 + tid, col, 32768)] = (__bf16)0.0f;   // pad rows
    const float* g0 = f1 + (size_t)(b*512 + j0) * 128;
    const float* g1 = f1 + (size_t)(b*512 + j1) * 128;
    const float* g2 = f1 + (size_t)(b*512 + j2) * 128;
    for (int c = tid; c < 128; c += 256)
        X[xpos(6 + c, col, 32768)] = (__bf16)(w0*g0[c] + w1*g1[c] + w2*g2[c]);
}

__global__ void permute_out_kernel(const float* __restrict__ Yh, float* __restrict__ out)
{   // Yh [40, B*8192] -> out [B, 40, 8192]
    int i = blockIdx.x * blockDim.x + threadIdx.x;
    if (i >= 40 * 32768) return;
    int d = i / 32768, col = i - d * 32768;
    int b = col >> 13, p = col & 8191;
    out[((size_t)b * 40 + d) * 8192 + p] = Yh[i];
}

// ---------------------------------------------------------------------------
// Host orchestration
// ---------------------------------------------------------------------------
struct LayerP { const float *cb, *sc, *bb; __bf16* Wp; int M, K32; };

extern "C" void kernel_launch(void* const* d_in, const int* in_sizes, int n_in,
                              void* d_out, int out_size, void* d_ws, size_t ws_size,
                              hipStream_t stream)
{
    (void)in_sizes; (void)n_in; (void)out_size; (void)ws_size;
    size_t cur = 0;
    auto allocB = [&](size_t bytes) {
        void* p = (char*)d_ws + cur;
        cur += (bytes + 255) / 256 * 256;
        return p;
    };
    auto allocF = [&](size_t elems) { return (float*)allocB(elems * 4); };
    auto allocH = [&](size_t elems) { return (__bf16*)allocB(elems * 2); };

    int q = 0;
    const float* in_xyz = (const float*)d_in[q++];
    auto packWeights = [&](const float* W, int cin, int cout) {
        LayerP L;
        L.M = cout;
        L.K32 = (cin + 31) / 32;
        int MtE = ((cout + 31) / 32) * 2;            // even # of 16-row tiles
        size_t total = (size_t)MtE * L.K32 * 512;
        L.Wp = allocH(total);
        pack_w_kernel<<<dim3((unsigned)((total + 255) / 256)), dim3(256), 0, stream>>>(
            W, L.Wp, cout, cin, L.K32, MtE);
        return L;
    };
    auto takeLayer = [&](int cin, int cout) {
        const float* W  = (const float*)d_in[q];
        LayerP L = packWeights(W, cin, cout);
        L.cb = (const float*)d_in[q+1];
        L.sc = (const float*)d_in[q+2];
        L.bb = (const float*)d_in[q+3];
        q += 4;
        return L;
    };

    LayerP sa1[3][3], sa2[3][3], sa3l[3], fp3l[2], fp2l[2], fp1l[3], head1, head2;
    const int sa1ch[3][3] = {{32,32,64},{64,64,128},{64,96,128}};
    for (int s = 0; s < 3; ++s) { int cin = 6; for (int l = 0; l < 3; ++l) { sa1[s][l] = takeLayer(cin, sa1ch[s][l]); cin = sa1ch[s][l]; } }
    const int sa2ch[3][3] = {{64,64,128},{128,128,256},{128,128,256}};
    for (int s = 0; s < 3; ++s) { int cin = 323; for (int l = 0; l < 3; ++l) { sa2[s][l] = takeLayer(cin, sa2ch[s][l]); cin = sa2ch[s][l]; } }
    { int cin = 643; const int ch[3]={256,512,1024}; for (int l=0;l<3;++l){ sa3l[l]=takeLayer(cin,ch[l]); cin=ch[l]; } }
    { int cin = 1664; const int ch[2]={256,256};     for (int l=0;l<2;++l){ fp3l[l]=takeLayer(cin,ch[l]); cin=ch[l]; } }
    { int cin = 576;  const int ch[2]={256,128};     for (int l=0;l<2;++l){ fp2l[l]=takeLayer(cin,ch[l]); cin=ch[l]; } }
    { int cin = 134;  const int ch[3]={128,128,128}; for (int l=0;l<3;++l){ fp1l[l]=takeLayer(cin,ch[l]); cin=ch[l]; } }
    head1 = takeLayer(128,128);
    {   // head2: (W2, b2) only
        const float* W2 = (const float*)d_in[q];
        const float* b2 = (const float*)d_in[q+1];
        q += 2;
        head2 = packWeights(W2, 128, 40);
        head2.cb = b2; head2.sc = nullptr; head2.bb = nullptr;
    }

    // ---- persistent activations / GEMM inputs
    float*  l1_xyz   = allocF((size_t)4*512*3);
    float*  l2_xyz   = allocF((size_t)4*128*3);
    float*  l1_feat  = allocF((size_t)4*512*320);       // point-major f32
    float*  l2_feat  = allocF((size_t)4*128*640);       // point-major f32
    __bf16* X_sa3p   = allocH((size_t)672*512);         // packed (643 -> 672)
    __bf16* X_fp3p   = allocH((size_t)1664*512);        // packed (exact)
    float*  l3       = allocF((size_t)4*1024);
    float*  l2new_pm = allocF((size_t)512*256);
    __bf16* X_fp2p   = allocH((size_t)576*2048);        // packed (exact)
    float*  l1new_pm = allocF((size_t)2048*128);
    __bf16* X_fp1p   = allocH((size_t)160*32768);       // packed (134 -> 160)
    float*  Yh       = allocF((size_t)40*32768);
    __bf16* Xgp      = allocH((size_t)352*16384);       // grouped input (max SA2 s3 per batch)
    float*  G0f      = allocF((size_t)128*65536);       // f32 out for pool/interp stages
    __bf16* G0p      = allocH((size_t)128*65536);       // packed ping
    __bf16* G1p      = allocH((size_t)128*65536);       // packed pong

    auto gemm = [&](const LayerP& L, const __bf16* Xp, float* Yf, __bf16* Yp,
                    int Ncols, int epi) {
        dim3 blk(128,1,1);
        int nt64 = Ncols / 64;                          // all Ncols are multiples of 64
        dim3 grd((unsigned)((nt64 + 3) / 4), (unsigned)((L.M + 31) / 32), 1);
        gemm_bn_kernel<<<grd, blk, 0, stream>>>(L.Wp, Xp, Yf, Yp, L.cb, L.sc, L.bb,
                                                L.M, L.K32, Ncols, epi);
    };

    // ---- FPS l0 -> 512 centers/batch
    fps_kernel<32><<<dim3(4), dim3(256), 0, stream>>>(in_xyz, 6, 8192, 512,
                                                      l1_xyz, (__bf16*)nullptr, 0);

    // ---- SA1 (multi-scale), per batch
    const float r1[3] = {0.1f, 0.2f, 0.4f};
    const int   K1[3] = {16, 32, 128};
    const int   C1off[3] = {0, 64, 192};
    for (int sc = 0; sc < 3; ++sc) {
        int K = K1[sc]; float r2v = r1[sc]*r1[sc];
        int Ncols = 512 * K;
        for (int b = 0; b < 4; ++b) {
            const float* xb = in_xyz + (size_t)b * 8192 * 6;
            group_kernel<<<dim3(512), dim3(256), 0, stream>>>(
                xb, 6, xb + 3, 6, 3, l1_xyz + (size_t)b*512*3, 8192, K, r2v,
                Xgp, Ncols, 32);
            gemm(sa1[sc][0], Xgp, nullptr, G0p, Ncols, 0);
            gemm(sa1[sc][1], G0p, nullptr, G1p, Ncols, 0);
            gemm(sa1[sc][2], G1p, G0f, nullptr, Ncols, 0);
            int D = sa1ch[sc][2];
            int tot = 512 * D;
            pool_kernel<<<dim3((tot+255)/256), dim3(256), 0, stream>>>(
                G0f, D, 512, K,
                l1_feat, 320, b*512, C1off[sc],
                X_fp2p, 2048, C1off[sc], b*512,
                (__bf16*)nullptr, 0, 0, 0);
        }
    }

    // ---- FPS l1 -> 128 centers/batch; xyz rows 0..2 of X_sa3 (packed)
    fps_kernel<2><<<dim3(4), dim3(256), 0, stream>>>(l1_xyz, 3, 512, 128,
                                                     l2_xyz, X_sa3p, 512);
    zero_rows_kernel<<<dim3((29*512+255)/256), dim3(256), 0, stream>>>(X_sa3p, 512, 643, 672);

    // ---- SA2 (multi-scale), per batch
    const float r2a[3] = {0.2f, 0.4f, 0.8f};
    const int   K2[3] = {32, 64, 128};
    const int   C2off[3] = {0, 128, 384};
    for (int sc = 0; sc < 3; ++sc) {
        int K = K2[sc]; float r2v = r2a[sc]*r2a[sc];
        int Ncols = 128 * K;
        for (int b = 0; b < 4; ++b) {
            group_kernel<<<dim3(128), dim3(256), 0, stream>>>(
                l1_xyz + (size_t)b*512*3, 3,
                l1_feat + (size_t)b*512*320, 320, 320,
                l2_xyz + (size_t)b*128*3, 512, K, r2v,
                Xgp, Ncols, 352);
            gemm(sa2[sc][0], Xgp, nullptr, G0p, Ncols, 0);
            gemm(sa2[sc][1], G0p, nullptr, G1p, Ncols, 0);
            gemm(sa2[sc][2], G1p, G0f, nullptr, Ncols, 0);
            int D = sa2ch[sc][2];
            int tot = 128 * D;
            pool_kernel<<<dim3((tot+255)/256), dim3(256), 0, stream>>>(
                G0f, D, 128, K,
                l2_feat, 640, b*128, C2off[sc],
                X_sa3p, 512, 3 + C2off[sc], b*128,
                X_fp3p, 512, C2off[sc], b*128);
        }
    }

    // ---- SA3 (group-all), N = 512 columns (4 batches x 128)
    gemm(sa3l[0], X_sa3p, nullptr, G0p, 512, 0);
    gemm(sa3l[1], G0p, nullptr, G1p, 512, 0);
    gemm(sa3l[2], G1p, G0f, nullptr, 512, 0);
    pool_all_kernel<<<dim3((4096+255)/256), dim3(256), 0, stream>>>(G0f, l3);

    // ---- FP3
    bcast_fp3_kernel<<<dim3((1024*512+255)/256), dim3(256), 0, stream>>>(l3, X_fp3p);
    gemm(fp3l[0], X_fp3p, nullptr, G0p, 512, 0);
    gemm(fp3l[1], G0p, G0f, nullptr, 512, 0);
    transpose_kernel<<<dim3((256*512+255)/256), dim3(256), 0, stream>>>(G0f, l2new_pm, 256, 512);

    // ---- FP2
    interp_fp2_kernel<<<dim3(2048), dim3(128), 0, stream>>>(l1_xyz, l2_xyz, l2new_pm, X_fp2p);
    gemm(fp2l[0], X_fp2p, nullptr, G0p, 2048, 0);
    gemm(fp2l[1], G0p, G0f, nullptr, 2048, 0);
    transpose_kernel<<<dim3((128*2048+255)/256), dim3(256), 0, stream>>>(G0f, l1new_pm, 128, 2048);

    // ---- FP1 + head
    build_fp1_kernel<<<dim3(32768), dim3(256), 0, stream>>>(in_xyz, l1_xyz, l1new_pm, X_fp1p);
    gemm(fp1l[0], X_fp1p, nullptr, G0p, 32768, 0);
    gemm(fp1l[1], G0p, nullptr, G1p, 32768, 0);
    gemm(fp1l[2], G1p, nullptr, G0p, 32768, 0);
    gemm(head1, G0p, nullptr, G1p, 32768, 0);
    gemm(head2, G1p, Yh, nullptr, 32768, 1);
    permute_out_kernel<<<dim3((40*32768+255)/256), dim3(256), 0, stream>>>(Yh, (float*)d_out);
}
